// MoEModel_30288109372026
// MI455X (gfx1250) — compile-verified, matrix operand order checked by reading, not run on hardware
//
#include <hip/hip_runtime.h>
#include <hip/hip_bf16.h>
#include <math.h>

typedef __attribute__((ext_vector_type(16))) _Float16 v16h;
typedef __attribute__((ext_vector_type(8)))  float    v8f;

#define BATCH 64
#define SEQ   512
#define DIM   768
#define WORDS 256
#define BW    (BATCH * WORDS)          // 16384 word rows (GEMM M)
#define HID   256
#define KDIM  (2 * DIM)                // 1536 (GEMM K)
#define NLAB  7
#define KB32  (KDIM / 32)              // 48 k-blocks
#define MT    (BW / 16)                // 1024 M-tiles
#define NT    (HID / 16)               // 16 N-tiles

// A/B 16-bit WMMA register layout (ISA 7.12.2):
//   lane%16 = row (A: M, B: N), half = lane/16,
//   halves e = 2v+p of the v16h map to k-offset:
//   koff = ((v&4)<<2) + (half<<3) + ((v&3)<<1) + p
// Packed storage: element16[ (tile*KB32 + kb)*32 + lane ][0..15] -> contiguous
// 32B per lane, coalesced b128 loads/stores.

// ---------------------------------------------------------------------------
// Kernel 1: segment-mean align (word_ids sorted -> binary search range).
// ---------------------------------------------------------------------------
__global__ void __launch_bounds__(256)
align_words_kernel(const float* __restrict__ hsub,   // (B,S,D)
                   const int*   __restrict__ wids,   // (B,S) sorted
                   float*       __restrict__ outW)   // (BW, D)
{
    const int bw = blockIdx.x;
    const int b  = bw >> 8;
    const int w  = bw & (WORDS - 1);
    const int* ids = wids + b * SEQ;

    int lo = 0, hi = SEQ;
    while (lo < hi) { int mid = (lo + hi) >> 1; if (ids[mid] <  w) lo = mid + 1; else hi = mid; }
    const int start = lo;
    hi = SEQ;
    while (lo < hi) { int mid = (lo + hi) >> 1; if (ids[mid] <= w) lo = mid + 1; else hi = mid; }
    const int end = lo;

    const int   cnt = end - start;
    const float inv = (cnt > 0) ? (1.0f / (float)cnt) : 0.0f;

    for (int d = threadIdx.x; d < DIM; d += blockDim.x) {
        float s = 0.0f;
        for (int i = start; i < end; ++i)
            s += hsub[((size_t)b * SEQ + i) * DIM + d];
        outW[(size_t)bw * DIM + d] = s * inv;
    }
}

// ---------------------------------------------------------------------------
// Kernel 2a: pack w1 (KDIM x HID f32, row-major) -> f16 B-register layout.
// One wave per (n-tile, k-block): 16*48 = 768 waves. Coalesced 32B stores.
// ---------------------------------------------------------------------------
__global__ void __launch_bounds__(256)
pack_w1_kernel(const float* __restrict__ w1, _Float16* __restrict__ bPack)
{
    const int lane = threadIdx.x & 31;
    const int gw   = blockIdx.x * 8 + (threadIdx.x >> 5);
    const int nt   = gw / KB32;
    const int kbI  = gw % KB32;
    const int kb   = kbI * 32;
    const int n    = nt * 16 + (lane & 15);
    const int half = lane >> 4;

    v16h bv;
#pragma unroll
    for (int v = 0; v < 8; ++v) {
        const int koff = ((v & 4) << 2) + (half << 3) + ((v & 3) << 1);
        const int k    = kb + koff;
        bv[2 * v]     = (_Float16)w1[(size_t)k       * HID + n];
        bv[2 * v + 1] = (_Float16)w1[(size_t)(k + 1) * HID + n];
    }
    *(v16h*)(bPack + ((size_t)(nt * KB32 + kbI) * 32 + lane) * 16) = bv;
}

// ---------------------------------------------------------------------------
// Kernel 2b: repack f32 word states -> f16 A-register layout (once, then the
// GEMM reads it with plain b128 loads; no gathers/converts in the hot loop).
// One wave per (m-tile, k-block): 1024*48 = 49152 waves.
// ---------------------------------------------------------------------------
__global__ void __launch_bounds__(256)
repack_A_kernel(const float* __restrict__ hingW, const float* __restrict__ robW,
                _Float16* __restrict__ aPack)
{
    const int lane = threadIdx.x & 31;
    const int gw   = blockIdx.x * 8 + (threadIdx.x >> 5);
    const int mt   = gw / KB32;
    const int kbI  = gw % KB32;
    const int kb   = kbI * 32;

    const size_t mrow = (size_t)(mt * 16 + (lane & 15));
    const int    half = lane >> 4;
    const float* src  = (kb < DIM) ? (hingW + mrow * DIM + kb)
                                   : (robW  + mrow * DIM + (kb - DIM));
    v16h av;
#pragma unroll
    for (int v = 0; v < 8; ++v) {
        const int koff = ((v & 4) << 2) + (half << 3) + ((v & 3) << 1);
        const float2 f = *(const float2*)(src + koff);
        av[2 * v]     = (_Float16)f.x;
        av[2 * v + 1] = (_Float16)f.y;
    }
    *(v16h*)(aPack + ((size_t)(mt * KB32 + kbI) * 32 + lane) * 16) = av;
}

// ---------------------------------------------------------------------------
// Kernel 3: router hidden GEMM, WMMA f32<-f16 16x16x32.
// Per-wave tile: 32(M) x 64(N) -> 2x4 accumulators, 8 WMMAs per k-step,
// 12 b128 loads per k-step, no converts. 2048 waves total.
// ---------------------------------------------------------------------------
__global__ void __launch_bounds__(256)
router_gemm_wmma_kernel(const _Float16* __restrict__ aPack,
                        const _Float16* __restrict__ bPack,
                        const float*    __restrict__ bias1,
                        float*          __restrict__ hidden)   // (BW, HID)
{
    const int lane = threadIdx.x & 31;
    const int gw   = blockIdx.x * 8 + (threadIdx.x >> 5);   // 2048 waves
    const int mp   = gw >> 2;                               // 512 M-pairs
    const int g    = gw & 3;                                // N group of 64
    const int mt0  = mp * 2;

    const int half  = lane >> 4;
    const int nbase = g * 64 + (lane & 15);

    const v16h* aV  = (const v16h*)aPack;
    const v16h* bV  = (const v16h*)bPack;
    const v16h* a0p = aV + (size_t)mt0 * KB32 * 32 + lane;
    const v16h* a1p = a0p + (size_t)KB32 * 32;
    const v16h* bp  = bV + (size_t)(g * 4) * KB32 * 32 + lane;
    const size_t BSTR = (size_t)KB32 * 32;                  // v16h stride per n-tile

    v8f acc[2][4] = {};

    for (int kbI = 0; kbI < KB32; ++kbI) {
        const v16h a0 = a0p[0];
        const v16h a1 = a1p[0];
        __builtin_prefetch(a0p + 32, 0, 3);   // next k-step of A
        const v16h b0 = bp[0];
        const v16h b1 = bp[BSTR];
        const v16h b2 = bp[2 * BSTR];
        const v16h b3 = bp[3 * BSTR];

        acc[0][0] = __builtin_amdgcn_wmma_f32_16x16x32_f16(false, a0, false, b0, (short)0, acc[0][0], false, false);
        acc[1][0] = __builtin_amdgcn_wmma_f32_16x16x32_f16(false, a1, false, b0, (short)0, acc[1][0], false, false);
        acc[0][1] = __builtin_amdgcn_wmma_f32_16x16x32_f16(false, a0, false, b1, (short)0, acc[0][1], false, false);
        acc[1][1] = __builtin_amdgcn_wmma_f32_16x16x32_f16(false, a1, false, b1, (short)0, acc[1][1], false, false);
        acc[0][2] = __builtin_amdgcn_wmma_f32_16x16x32_f16(false, a0, false, b2, (short)0, acc[0][2], false, false);
        acc[1][2] = __builtin_amdgcn_wmma_f32_16x16x32_f16(false, a1, false, b2, (short)0, acc[1][2], false, false);
        acc[0][3] = __builtin_amdgcn_wmma_f32_16x16x32_f16(false, a0, false, b3, (short)0, acc[0][3], false, false);
        acc[1][3] = __builtin_amdgcn_wmma_f32_16x16x32_f16(false, a1, false, b3, (short)0, acc[1][3], false, false);

        a0p += 32; a1p += 32; bp += 32;
    }

    // epilogue: bias + relu, scatter per C/D layout (VGPR r -> M = r + half*8)
#pragma unroll
    for (int i = 0; i < 2; ++i) {
#pragma unroll
        for (int j = 0; j < 4; ++j) {
            const int   ncol = nbase + j * 16;
            const float bb   = bias1[ncol];
#pragma unroll
            for (int r = 0; r < 8; ++r) {
                const int m = (mt0 + i) * 16 + r + half * 8;
                hidden[(size_t)m * HID + ncol] = fmaxf(acc[i][j][r] + bb, 0.0f);
            }
        }
    }
}

// ---------------------------------------------------------------------------
// Kernel 4: fused tail per word row (alpha, blend, 7-wide head, l2 norm).
// ---------------------------------------------------------------------------
__global__ void __launch_bounds__(256)
finalize_kernel(const float* __restrict__ hingW,
                const float* __restrict__ robW,
                const float* __restrict__ hidden,
                const float* __restrict__ w2,      // (HID)
                const float* __restrict__ b2,      // (1)
                const float* __restrict__ head_w,  // (DIM, NLAB)
                const float* __restrict__ head_b,  // (NLAB)
                float*       __restrict__ out)     // logits | alpha | l2
{
    const int bw = blockIdx.x;
    const int t  = threadIdx.x;
    __shared__ float red[256];
    __shared__ float alpha_sh;

    red[t] = hidden[(size_t)bw * HID + t] * w2[t];
    __syncthreads();
#pragma unroll
    for (int s = 128; s > 0; s >>= 1) {
        if (t < s) red[t] += red[t + s];
        __syncthreads();
    }
    if (t == 0) alpha_sh = 1.0f / (1.0f + expf(-(red[0] + b2[0])));
    __syncthreads();
    const float alpha = alpha_sh;

    float part[NLAB + 1];
#pragma unroll
    for (int c = 0; c < NLAB + 1; ++c) part[c] = 0.0f;

    for (int d = t; d < DIM; d += 256) {
        const float hh = hingW[(size_t)bw * DIM + d];
        const float hr = robW [(size_t)bw * DIM + d];
        const float bl = alpha * hh + (1.0f - alpha) * hr;
        const float df = hh - hr;
        part[NLAB] += df * df;
#pragma unroll
        for (int l = 0; l < NLAB; ++l)
            part[l] += bl * head_w[d * NLAB + l];
    }

#pragma unroll
    for (int c = 0; c < NLAB + 1; ++c) {
        __syncthreads();
        red[t] = part[c];
        __syncthreads();
#pragma unroll
        for (int s = 128; s > 0; s >>= 1) {
            if (t < s) red[t] += red[t + s];
            __syncthreads();
        }
        if (t == 0) {
            if (c < NLAB)
                out[(size_t)bw * NLAB + c] = red[0] + head_b[c];
            else
                out[(size_t)BW * NLAB + BW + bw] = sqrtf(red[0]);
        }
    }
    if (t == 0) out[(size_t)BW * NLAB + bw] = alpha;
}

// ---------------------------------------------------------------------------
extern "C" void kernel_launch(void* const* d_in, const int* in_sizes, int n_in,
                              void* d_out, int out_size, void* d_ws, size_t ws_size,
                              hipStream_t stream) {
    const float* h_hing_sub = (const float*)d_in[0];
    const float* h_rob_sub  = (const float*)d_in[1];
    const int*   hing_ids   = (const int*)d_in[2];
    const int*   rob_ids    = (const int*)d_in[3];
    // d_in[4] = num_words: unused by the reference computation
    const float* router_w1  = (const float*)d_in[5];
    const float* router_b1  = (const float*)d_in[6];
    const float* router_w2  = (const float*)d_in[7];
    const float* router_b2  = (const float*)d_in[8];
    const float* head_w     = (const float*)d_in[9];
    const float* head_b     = (const float*)d_in[10];

    float* out    = (float*)d_out;
    float* ws     = (float*)d_ws;
    float*     hingW  = ws;                               // BW*DIM f32   (48 MB)
    float*     robW   = hingW + (size_t)BW * DIM;         // BW*DIM f32   (48 MB)
    float*     hidden = robW  + (size_t)BW * DIM;         // BW*HID f32   (16 MB)
    _Float16*  aPack  = (_Float16*)(hidden + (size_t)BW * HID);  // BW*KDIM f16 (48 MB)
    _Float16*  bPack  = aPack + (size_t)BW * KDIM;        // KDIM*HID f16 (0.75 MB)

    align_words_kernel<<<BW, 256, 0, stream>>>(h_hing_sub, hing_ids, hingW);
    align_words_kernel<<<BW, 256, 0, stream>>>(h_rob_sub,  rob_ids,  robW);

    pack_w1_kernel<<<(NT * KB32) / 8, 256, 0, stream>>>(router_w1, bPack);
    repack_A_kernel<<<(MT * KB32) / 8, 256, 0, stream>>>(hingW, robW, aPack);

    // 512 M-pairs * 4 N-groups = 2048 waves; 8 waves/block -> 256 blocks
    router_gemm_wmma_kernel<<<256, 256, 0, stream>>>(aPack, bPack, router_b1, hidden);

    finalize_kernel<<<BW, 256, 0, stream>>>(hingW, robW, hidden,
                                            router_w2, router_b2, head_w, head_b, out);
}
